// ComplEx_72576357368236
// MI455X (gfx1250) — compile-verified, matrix operand order checked by reading, not run on hardware
//
#include <hip/hip_runtime.h>
#include <stdint.h>

// ComplEx scoring: out[b,e] = tmp1[b,:]·E_im[e,:] + tmp2[b,:]·E_re[e,:] + mask[e]
// Split-f16 (hi+lo) WMMA GEMM. E panel staged to LDS via the Tensor Data
// Mover (tensor_load_to_lds, double-buffered, TENSORcnt) with a per-lane
// global_load_async_to_lds fallback; cooperative f32->f16 conversion in LDS;
// fragments read via ds_load_b128 in native WMMA layout.

typedef __attribute__((ext_vector_type(16))) _Float16 v16h;
typedef __attribute__((ext_vector_type(8)))  _Float16 v8h;
typedef __attribute__((ext_vector_type(2)))  _Float16 v2h;
typedef __attribute__((ext_vector_type(8)))  float    v8f;
typedef __attribute__((ext_vector_type(4)))  unsigned int v4u;
typedef __attribute__((ext_vector_type(8)))  int      v8i;
typedef __attribute__((ext_vector_type(4)))  int      v4i;

#define BATCH   1024
#define DIMK    512
#define NENT    100000
#define NRELH   500

#define MT      128      // block M tile (4 M-waves x 32)
#define NT      64       // block N tile (2 N-waves x 32)
#define KC      32       // K chunk per WMMA step
#define NCHUNK  32       // 2 phases * (512/32)

#if __has_builtin(__builtin_amdgcn_tensor_load_to_lds)
#define USE_TDM 1
#else
#define USE_TDM 0
#endif

// ------------------------------------------------------------------
// Kernel 1: gather + complex rotation + f16 hi/lo split of tmp1/tmp2.
// A panels (1024 x 512) f16, row-major. 4 MB total in ws (L2 resident).
// ------------------------------------------------------------------
__global__ __launch_bounds__(256)
void complex_prep(const int* __restrict__ h, const int* __restrict__ r,
                  const float* __restrict__ E_im, const float* __restrict__ E_re,
                  const float* __restrict__ R_im, const float* __restrict__ R_re,
                  _Float16* __restrict__ A1hi, _Float16* __restrict__ A1lo,
                  _Float16* __restrict__ A2hi, _Float16* __restrict__ A2lo) {
  int idx = blockIdx.x * blockDim.x + threadIdx.x;   // 1024*512 threads
  int b = idx >> 9;
  int d = idx & (DIMK - 1);
  int hh = h[b];
  int rr = r[b];
  bool inv = rr >= NRELH;
  int reff = inv ? rr - NRELH : rr;
  float him = E_im[(size_t)hh * DIMK + d];
  float hre = E_re[(size_t)hh * DIMK + d];
  float rim = R_im[(size_t)reff * DIMK + d];
  float rre = R_re[(size_t)reff * DIMK + d];
  float t1 = him * rre + (inv ? -(hre * rim) : (hre * rim));
  float t2 = inv ? (him * rim + hre * rre) : (hre * rre - him * rim);
  _Float16 t1h = (_Float16)t1;
  _Float16 t2h = (_Float16)t2;
  A1hi[idx] = t1h;
  A1lo[idx] = (_Float16)(t1 - (float)t1h);
  A2hi[idx] = t2h;
  A2lo[idx] = (_Float16)(t2 - (float)t2h);
}

// WMMA 16-bit fragment layout (ISA 7.12.2), identical for A and B:
// lane: elements 0..7  = K[ (lane>>4)*8      .. +7 ]
//       elements 8..15 = K[ 16 + (lane>>4)*8 .. +7 ]   (K-contiguous runs)
__device__ __forceinline__ v16h combine8(v8h a, v8h b) {
  return __builtin_shufflevector(a, b, 0, 1, 2, 3, 4, 5, 6, 7,
                                       8, 9, 10, 11, 12, 13, 14, 15);
}

__device__ __forceinline__ v16h load_frag_f16(const _Float16* __restrict__ base,
                                              int row, int k, int lane) {
  const _Float16* p = base + (size_t)row * DIMK + k + ((lane >> 4) << 3);
  return combine8(*(const v8h*)(p), *(const v8h*)(p + 16));
}

#if USE_TDM
// One TDM descriptor moves the whole 64-row x 32-float tile (stride 512 f32)
// into LDS. tensor_dim1 = remaining rows -> OOB rows read as zero (their
// output columns are never stored). D# fields per ISA 8.3/8.4.
// This toolchain's builtin is the 6-arg form:
//   (v4u group0, v8i group1, v4i, v4i, v8i, i32 cpol)
// groups beyond 0/1 describe dims 2..4 -> zero for a 2D tile.
__device__ __forceinline__ void tdm_issue(const float* __restrict__ E,
                                          int nblk, int k, unsigned lds_off) {
  unsigned long long ga =
      (unsigned long long)(uintptr_t)(E + (size_t)nblk * DIMK + k);
  unsigned dim1 = (unsigned)(NENT - nblk);          // remaining rows
  v4u g0 = { 1u,                                    // count=1, user desc
             lds_off,                               // lds_addr (bytes)
             (unsigned)ga,                          // global_addr[31:0]
             (((unsigned)(ga >> 32)) & 0x01FFFFFFu) | 0x80000000u }; // type=2
  v8i g1 = { (int)0x00020000u,                      // data_size=4B
             (int)(32u << 16),                      // tensor_dim0 = 32
             (int)((dim1 & 0xFFFFu) << 16),         // tensor_dim1 lo16
             (int)(((dim1 >> 16) & 0xFFFFu) | (32u << 16)), // dim1 hi | tile_dim0=32
             64,                                    // tile_dim1 = 64 rows
             512,                                   // tensor_dim0_stride lo32
             0, 0 };
  v4i gz4 = { 0, 0, 0, 0 };                         // dims 2..4 unused
  v8i gz8 = { 0, 0, 0, 0, 0, 0, 0, 0 };
  __builtin_amdgcn_tensor_load_to_lds(g0, g1, gz4, gz4, gz8, 0);
}
#endif

// ------------------------------------------------------------------
// Kernel 2: GEMM. Block = 8 wave32 (4M x 2N), wave tile 32x32,
// block tile 128 x 64. E tile (64 x 32 f32 = 8KB) DMA'd to LDS,
// converted once per block to f16 hi/lo, fragments read via ds_load_b128.
// 3-product split-f16 WMMA: hi*hi + lo*hi + hi*lo (~f32 accuracy).
// ------------------------------------------------------------------
__global__ __launch_bounds__(256)
void complex_gemm(const float* __restrict__ E_im, const float* __restrict__ E_re,
                  const float* __restrict__ mask,
                  const _Float16* __restrict__ A1hi, const _Float16* __restrict__ A1lo,
                  const _Float16* __restrict__ A2hi, const _Float16* __restrict__ A2lo,
                  float* __restrict__ out) {
  __shared__ float    LF32[2][NT * KC];   // 2 x 8 KB f32 staging
  __shared__ _Float16 LHI [2][NT * KC];   // 2 x 4 KB
  __shared__ _Float16 LLO [2][NT * KC];   // 2 x 4 KB

  const int t    = threadIdx.x;
  const int lane = t & 31;
  const int wave = t >> 5;
  const int wm   = wave & 3;              // 4 M-waves
  const int wn   = wave >> 2;             // 2 N-waves
  const int m0   = blockIdx.x * MT + wm * 32;
  const int nblk = blockIdx.y * NT;
  const int lr   = lane & 15;
  const int kh   = (lane >> 4) << 3;      // per-lane K sub-offset

  v8f c[2][2] = {};

#if USE_TDM
  auto issue = [&](int cidx) {            // wave-0 only; one DMA per chunk
    const int phase = cidx >> 4;
    const int k     = (cidx & 15) * KC;
    const int buf   = cidx & 1;
    tdm_issue(phase ? E_re : E_im, nblk, k,
              (unsigned)(uintptr_t)(&LF32[buf][0]));
  };
  if (wave == 0) issue(0);
#else
  auto issue = [&](int cidx) {            // per-lane async copies (fallback)
    const int   phase = cidx >> 4;
    const int   k     = (cidx & 15) * KC;
    const int   buf   = cidx & 1;
    const float* E    = phase ? E_re : E_im;
#pragma unroll
    for (int i = 0; i < 2; ++i) {
      int cq  = t + i * 256;
      int n   = cq >> 3;
      int kk  = (cq & 7) << 2;
      int row = nblk + n;
      if (row > NENT - 1) row = NENT - 1;
      const float* g = E + (size_t)row * DIMK + k + kk;
      unsigned lds = (unsigned)(uintptr_t)(&LF32[buf][n * KC + kk]);
      asm volatile("global_load_async_to_lds_b128 %0, %1, off"
                   :: "v"(lds), "v"(g) : "memory");
    }
  };
  issue(0);
#endif

  for (int cidx = 0; cidx < NCHUNK; ++cidx) {
    const int buf   = cidx & 1;
    const int phase = cidx >> 4;
    const int k     = (cidx & 15) * KC;

#if USE_TDM
    if (wave == 0) {
      if (cidx + 1 < NCHUNK) {
        issue(cidx + 1);
        __builtin_amdgcn_s_wait_tensorcnt(1);   // chunk cidx retired
      } else {
        __builtin_amdgcn_s_wait_tensorcnt(0);
      }
    }
#else
    if (cidx + 1 < NCHUNK) {
      issue(cidx + 1);
      asm volatile("s_wait_asynccnt 0x2" ::: "memory");
    } else {
      asm volatile("s_wait_asynccnt 0x0" ::: "memory");
    }
#endif
    __syncthreads();                      // chunk cidx visible to all waves

    // Cooperative convert: 2048 elems, 8 per thread, packed v2h stores.
#pragma unroll
    for (int i = 0; i < 4; ++i) {
      int j = (t + i * 256) * 2;
      float f0 = LF32[buf][j];
      float f1 = LF32[buf][j + 1];
      _Float16 h0 = (_Float16)f0, h1 = (_Float16)f1;
      v2h hv = {h0, h1};
      v2h lv = {(_Float16)(f0 - (float)h0), (_Float16)(f1 - (float)h1)};
      *(v2h*)&LHI[buf][j] = hv;
      *(v2h*)&LLO[buf][j] = lv;
    }
    __syncthreads();                      // f16 tile ready

    const _Float16* Ahi = phase ? A2hi : A1hi;
    const _Float16* Alo = phase ? A2lo : A1lo;

    v16h ahi[2], alo[2], bhi[2], blo[2];
#pragma unroll
    for (int fm = 0; fm < 2; ++fm) {
      int row = m0 + fm * 16 + lr;
      ahi[fm] = load_frag_f16(Ahi, row, k, lane);
      alo[fm] = load_frag_f16(Alo, row, k, lane);
    }
#pragma unroll
    for (int fn = 0; fn < 2; ++fn) {
      int nn = wn * 32 + fn * 16 + lr;    // column within N tile
      const _Float16* ph = &LHI[buf][nn * KC + kh];
      const _Float16* pl = &LLO[buf][nn * KC + kh];
      bhi[fn] = combine8(*(const v8h*)ph, *(const v8h*)(ph + 16));
      blo[fn] = combine8(*(const v8h*)pl, *(const v8h*)(pl + 16));
    }

#pragma unroll
    for (int fm = 0; fm < 2; ++fm)
#pragma unroll
      for (int fn = 0; fn < 2; ++fn) {
        c[fm][fn] = __builtin_amdgcn_wmma_f32_16x16x32_f16(
            false, ahi[fm], false, bhi[fn], (short)0, c[fm][fn], false, false);
        c[fm][fn] = __builtin_amdgcn_wmma_f32_16x16x32_f16(
            false, alo[fm], false, bhi[fn], (short)0, c[fm][fn], false, false);
        c[fm][fn] = __builtin_amdgcn_wmma_f32_16x16x32_f16(
            false, ahi[fm], false, blo[fn], (short)0, c[fm][fn], false, false);
      }
  }

  // C fragment: lane = N (mod 16), VGPR j -> M = j + 8*(lane>>4)
  const int lh = lane >> 4;
#pragma unroll
  for (int fn = 0; fn < 2; ++fn) {
    int n = nblk + wn * 32 + fn * 16 + lr;
    if (n < NENT) {
      float mv = mask[n];
#pragma unroll
      for (int fm = 0; fm < 2; ++fm) {
#pragma unroll
        for (int j = 0; j < 8; ++j) {
          int m = m0 + fm * 16 + lh * 8 + j;
          out[(size_t)m * NENT + n] = c[fm][fn][j] + mv;
        }
      }
    }
  }
}

// ------------------------------------------------------------------
extern "C" void kernel_launch(void* const* d_in, const int* in_sizes, int n_in,
                              void* d_out, int out_size, void* d_ws, size_t ws_size,
                              hipStream_t stream) {
  const int*   h    = (const int*)d_in[0];
  const int*   r    = (const int*)d_in[1];
  const float* E_im = (const float*)d_in[2];
  const float* E_re = (const float*)d_in[3];
  const float* R_im = (const float*)d_in[4];
  const float* R_re = (const float*)d_in[5];
  const float* mask = (const float*)d_in[6];
  float* out = (float*)d_out;

  // ws layout: 4 x (1024*512) f16 panels = 4 MB
  _Float16* A1hi = (_Float16*)d_ws;
  _Float16* A1lo = A1hi + (size_t)BATCH * DIMK;
  _Float16* A2hi = A1lo + (size_t)BATCH * DIMK;
  _Float16* A2lo = A2hi + (size_t)BATCH * DIMK;

  complex_prep<<<(BATCH * DIMK) / 256, 256, 0, stream>>>(
      h, r, E_im, E_re, R_im, R_re, A1hi, A1lo, A2hi, A2lo);

  dim3 grid(BATCH / MT, (NENT + NT - 1) / NT);   // 8 x 1563
  complex_gemm<<<grid, 256, 0, stream>>>(
      E_im, E_re, mask, A1hi, A1lo, A2hi, A2lo, out);
}
